// MoEClassifier_61675730370568
// MI455X (gfx1250) — matrix operand order repaired and checked by reading, not compiled, not run on hardware
//
#include <hip/hip_runtime.h>
#include <stdint.h>
#include <stddef.h>

// ---------------- problem constants ----------------
constexpr int BATCH = 16384;
constexpr int DIMD  = 1024;
constexpr int HID   = 4096;
constexpr int NCLS  = 512;
constexpr int NEXP  = 8;
constexpr int SLOTS = 33792;   // sum_e roundup(count_e,128) <= 32768 + 8*127 -> pad to 264*128
constexpr int NTILES = 264;

typedef __bf16 bf16;
typedef __attribute__((ext_vector_type(16))) __bf16 v16bf;
typedef __attribute__((ext_vector_type(8)))  __bf16 bf16x8;
typedef __attribute__((ext_vector_type(4)))  __bf16 bf16x4;
typedef __attribute__((ext_vector_type(8)))  float  v8f;

// ---------------- CDNA5 async global->LDS helpers ----------------
__device__ __forceinline__ void async_load_b128(unsigned lds_off, const void* g) {
  // VDST = LDS byte address VGPR, VADDR = 64-bit global address, saddr = off
  asm volatile("global_load_async_to_lds_b128 %0, %1, off"
               :: "v"(lds_off), "v"((unsigned long long)(uintptr_t)g)
               : "memory");
}
__device__ __forceinline__ void wait_async0() {
  asm volatile("s_wait_asynccnt 0" ::: "memory");
}

__device__ __forceinline__ v16bf pack16(bf16x8 lo, bf16x8 hi) {
  return __builtin_shufflevector(lo, hi, 0,1,2,3,4,5,6,7,8,9,10,11,12,13,14,15);
}

// ---------------- 1) router: fp32 logits, top-2, softmax ----------------
__global__ __launch_bounds__(256) void moe_router_kernel(
    const float* __restrict__ x, const float* __restrict__ gw, const float* __restrict__ gb,
    float* __restrict__ logits, int* __restrict__ tokexp, float* __restrict__ tokp)
{
  const int t    = blockIdx.x * 8 + (threadIdx.x >> 5);
  const int lane = threadIdx.x & 31;
  const float* xr = x + (long)t * DIMD;
  float a[8] = {0.f,0.f,0.f,0.f,0.f,0.f,0.f,0.f};
  for (int d = lane; d < DIMD; d += 32) {
    const float xv = xr[d];
    const float4* g4 = (const float4*)(gw + d * NEXP);
    const float4 g0 = g4[0], g1 = g4[1];
    a[0] += xv * g0.x; a[1] += xv * g0.y; a[2] += xv * g0.z; a[3] += xv * g0.w;
    a[4] += xv * g1.x; a[5] += xv * g1.y; a[6] += xv * g1.z; a[7] += xv * g1.w;
  }
  #pragma unroll
  for (int e = 0; e < 8; ++e) {
    #pragma unroll
    for (int m = 16; m > 0; m >>= 1) a[e] += __shfl_xor(a[e], m);
    a[e] += gb[e];
  }
  if (lane == 0) {
    #pragma unroll
    for (int e = 0; e < 8; ++e) logits[(long)t * NEXP + e] = a[e];
    int i0 = 0; float v0 = a[0];
    #pragma unroll
    for (int e = 1; e < 8; ++e) if (a[e] > v0) { v0 = a[e]; i0 = e; }
    int i1 = (i0 == 0) ? 1 : 0; float v1 = a[i1];
    #pragma unroll
    for (int e = 0; e < 8; ++e) if (e != i0 && a[e] > v1) { v1 = a[e]; i1 = e; }
    const float p1 = expf(v1 - v0);
    const float s  = 1.f + p1;
    tokexp[2 * t] = i0; tokexp[2 * t + 1] = i1;
    tokp[2 * t] = 1.f / s; tokp[2 * t + 1] = p1 / s;
  }
}

// ---------------- 2) bucketing ----------------
__global__ __launch_bounds__(256) void moe_init_kernel(int* token_of, int* cnt, int* cursor)
{
  const int i = blockIdx.x * 256 + threadIdx.x;
  if (i < SLOTS) token_of[i] = -1;
  if (i < NEXP) { cnt[i] = 0; cursor[i] = 0; }
}

__global__ __launch_bounds__(256) void moe_count_kernel(const int* __restrict__ tokexp, int* cnt)
{
  const int t = blockIdx.x * 256 + threadIdx.x;
  if (t < BATCH) {
    atomicAdd(&cnt[tokexp[2 * t]], 1);
    atomicAdd(&cnt[tokexp[2 * t + 1]], 1);
  }
}

__global__ void moe_scan_kernel(const int* __restrict__ cnt, int* __restrict__ off,
                                int* __restrict__ tile_expert)
{
  if (threadIdx.x == 0 && blockIdx.x == 0) {
    int o = 0;
    for (int e = 0; e < NEXP; ++e) { off[e] = o; o += (cnt[e] + 127) & ~127; }
    off[NEXP] = o;
    for (int tlt = 0; tlt < NTILES; ++tlt) {
      const int r = tlt * 128;
      int e = 0;
      while (e < NEXP - 1 && r >= off[e + 1]) ++e;
      tile_expert[tlt] = e;
    }
  }
}

__global__ __launch_bounds__(256) void moe_fill_kernel(
    const int* __restrict__ tokexp, const int* __restrict__ off,
    int* __restrict__ cursor, int* __restrict__ token_of, int* __restrict__ slot_of)
{
  const int t = blockIdx.x * 256 + threadIdx.x;
  if (t < BATCH) {
    #pragma unroll
    for (int k = 0; k < 2; ++k) {
      const int e = tokexp[2 * t + k];
      const int p = atomicAdd(&cursor[e], 1);
      const int s = off[e] + p;
      token_of[s] = t;
      slot_of[2 * t + k] = s;
    }
  }
}

// ---------------- 3) precision prep ----------------
__global__ __launch_bounds__(256) void moe_convx_kernel(const float* __restrict__ x, bf16* __restrict__ xb)
{
  const size_t i = ((size_t)blockIdx.x * 256 + threadIdx.x) * 4;
  const float4 v = *(const float4*)(x + i);
  bf16x4 r; r[0] = (bf16)v.x; r[1] = (bf16)v.y; r[2] = (bf16)v.z; r[3] = (bf16)v.w;
  *(bf16x4*)(xb + i) = r;
}

// src: [E][R][Cc] fp32 -> dst: [E][Cc][R] bf16 (LDS-tiled transpose)
__global__ __launch_bounds__(256) void moe_transpose_conv_kernel(
    const float* __restrict__ src, bf16* __restrict__ dst, int R, int Cc)
{
  __shared__ float tile[32][33];
  const float* s = src + (size_t)blockIdx.z * R * Cc;
  bf16* d = dst + (size_t)blockIdx.z * R * Cc;
  const int c0 = blockIdx.x * 32, r0 = blockIdx.y * 32;
  const int tx = threadIdx.x & 31, ty = threadIdx.x >> 5;
  #pragma unroll
  for (int i = 0; i < 4; ++i)
    tile[ty + 8 * i][tx] = s[(size_t)(r0 + ty + 8 * i) * Cc + c0 + tx];
  __syncthreads();
  #pragma unroll
  for (int i = 0; i < 4; ++i)
    d[(size_t)(c0 + ty + 8 * i) * R + r0 + tx] = (bf16)tile[tx][ty + 8 * i];
}

// ---------------- 4/6) bf16 WMMA GEMM: 128x128 tile, K step 32 ----------------
// A: [rows][KDIM] bf16 (optionally gathered via token_of); Bt: [E][NDIM][KDIM] bf16 (pre-transposed)
// Out: STORE_BF16 ? bf16[rows][NDIM] : float[rows][NDIM]; bias: [E][NDIM]
template<int KDIM, int NDIM, bool GATHER, bool STORE_BF16>
__global__ __launch_bounds__(256) void moe_gemm_kernel(
    const bf16* __restrict__ A, const bf16* __restrict__ Bt,
    const float* __restrict__ bias, const int* __restrict__ token_of,
    const int* __restrict__ tile_expert, void* __restrict__ Out)
{
  constexpr int NK = KDIM / 32;
  constexpr unsigned ABUF = 128 * 40 * 2;        // bytes per A buffer (stride 40 bf16, padded)
  __shared__ bf16 sA[2][128 * 40];

  const int rt   = blockIdx.y;
  const int e    = tile_expert[rt];
  const int row0 = rt * 128;
  const int n0   = blockIdx.x * 128;
  const int tid  = threadIdx.x;
  const int lane = tid & 31;
  const int wid  = tid >> 5;
  const int wm   = wid & 3;        // 4 row groups of 32
  const int wn   = wid >> 2;       // 2 col groups of 64
  const int half = lane >> 4;
  const int l16  = lane & 15;

  // A staging ownership: row = tid>>1, 16-element segment = tid&1 (32B via 2x async b128)
  const int arow = tid >> 1;
  const int aseg = tid & 1;
  long asrcrow;
  if (GATHER) { int t = token_of[row0 + arow]; if (t < 0) t = 0; asrcrow = (long)t * KDIM; }
  else        { asrcrow = (long)(row0 + arow) * KDIM; }
  const bf16* agp = A + asrcrow + aseg * 16;
  const unsigned lds0 = (unsigned)(size_t)&sA[0][arow * 40 + aseg * 16];

  v8f acc[2][4];
  #pragma unroll
  for (int i = 0; i < 2; ++i)
    #pragma unroll
    for (int j = 0; j < 4; ++j)
      acc[i][j] = (v8f){0.f,0.f,0.f,0.f,0.f,0.f,0.f,0.f};

  // prologue: fill buffer 0
  async_load_b128(lds0,      agp);
  async_load_b128(lds0 + 16, agp + 8);

  const bf16* bbase = Bt + (long)e * NDIM * KDIM;

  for (int kk = 0; kk < NK; ++kk) {
    const int cur = kk & 1;
    wait_async0();
    __syncthreads();                           // current buffer complete for all waves
    if (kk + 1 < NK) {                         // prefetch next buffer (overlaps compute)
      const bf16* g = agp + (kk + 1) * 32;
      const unsigned l = lds0 + (unsigned)(1 - cur) * ABUF;
      async_load_b128(l,      g);
      async_load_b128(l + 16, g + 8);
    }
    // A fragments from LDS (contiguous layout: K=16h+j, matching B)
    v16bf afr[2];
    #pragma unroll
    for (int mt = 0; mt < 2; ++mt) {
      const bf16* ap = &sA[cur][(wm * 32 + mt * 16 + l16) * 40 + 16 * half];
      const bf16x8 lo = *(const bf16x8*)ap;
      const bf16x8 hi = *(const bf16x8*)(ap + 8);
      afr[mt] = pack16(lo, hi);
    }
    // B fragments straight from transposed global (lane: n=l16, K=16h+j contiguous)
    #pragma unroll
    for (int nt = 0; nt < 4; ++nt) {
      const int n = n0 + wn * 64 + nt * 16 + l16;
      const bf16* bp = bbase + (long)n * KDIM + kk * 32 + 16 * half;
      const bf16x8 lo = *(const bf16x8*)bp;
      const bf16x8 hi = *(const bf16x8*)(bp + 8);
      const v16bf bfr = pack16(lo, hi);
      acc[0][nt] = __builtin_amdgcn_wmma_f32_16x16x32_bf16(false, afr[0], false, bfr,
                                                           (short)0, acc[0][nt], false, false);
      acc[1][nt] = __builtin_amdgcn_wmma_f32_16x16x32_bf16(false, afr[1], false, bfr,
                                                           (short)0, acc[1][nt], false, false);
    }
    __syncthreads();                           // all reads of cur done before overwrite
  }

  // epilogue: + bias, store (C layout: lane l, vgpr v -> M=v+8*half, N=l16)
  #pragma unroll
  for (int mt = 0; mt < 2; ++mt)
    #pragma unroll
    for (int nt = 0; nt < 4; ++nt) {
      const int col = n0 + wn * 64 + nt * 16 + l16;
      const float bv = bias[e * NDIM + col];
      #pragma unroll
      for (int v = 0; v < 8; ++v) {
        const int row = row0 + wm * 32 + mt * 16 + 8 * half + v;
        const float val = acc[mt][nt][v] + bv;
        if (STORE_BF16) ((bf16*) Out)[(long)row * NDIM + col] = (bf16)val;
        else            ((float*)Out)[(long)row * NDIM + col] = val;
      }
    }
}

// ---------------- 5) LayerNorm + exact-erf GELU (in place), one wave per row ----------------
__global__ __launch_bounds__(256) void moe_ln_gelu_kernel(
    bf16* __restrict__ h, const int* __restrict__ tile_expert,
    const float* __restrict__ ln_g, const float* __restrict__ ln_b)
{
  const int row  = blockIdx.x * 8 + (threadIdx.x >> 5);
  const int lane = threadIdx.x & 31;
  const int e    = tile_expert[row >> 7];
  bf16* hp = h + (long)row * HID;
  float s = 0.f, sq = 0.f;
  #pragma unroll 4
  for (int i = 0; i < 16; ++i) {
    const bf16x8 v = *(const bf16x8*)(hp + i * 256 + lane * 8);
    #pragma unroll
    for (int j = 0; j < 8; ++j) { const float f = (float)v[j]; s += f; sq += f * f; }
  }
  #pragma unroll
  for (int m = 16; m > 0; m >>= 1) { s += __shfl_xor(s, m); sq += __shfl_xor(sq, m); }
  const float mu = s * (1.f / HID);
  const float rs = rsqrtf(sq * (1.f / HID) - mu * mu + 1e-5f);
  const float* g = ln_g + (long)e * HID;
  const float* b = ln_b + (long)e * HID;
  #pragma unroll 2
  for (int i = 0; i < 16; ++i) {
    const int k = i * 256 + lane * 8;
    const bf16x8 v = *(const bf16x8*)(hp + k);
    const float4 g0 = *(const float4*)(g + k), g1 = *(const float4*)(g + k + 4);
    const float4 b0 = *(const float4*)(b + k), b1 = *(const float4*)(b + k + 4);
    const float gg[8] = {g0.x,g0.y,g0.z,g0.w,g1.x,g1.y,g1.z,g1.w};
    const float bb[8] = {b0.x,b0.y,b0.z,b0.w,b1.x,b1.y,b1.z,b1.w};
    bf16x8 o;
    #pragma unroll
    for (int j = 0; j < 8; ++j) {
      const float t = ((float)v[j] - mu) * rs * gg[j] + bb[j];
      const float a = 0.5f * t * (1.f + erff(t * 0.70710678118654752f));
      o[j] = (bf16)a;
    }
    *(bf16x8*)(hp + k) = o;
  }
}

// ---------------- 7) combine: out[t] = p0*part[s0] + p1*part[s1] ----------------
__global__ __launch_bounds__(256) void moe_combine_kernel(
    const float* __restrict__ part, const int* __restrict__ slot_of,
    const float* __restrict__ tokp, float* __restrict__ out)
{
  const int gid = blockIdx.x * 256 + threadIdx.x;     // one float4 each
  const int t = gid >> 7;                             // 512/4 = 128 groups per token
  const int c = (gid & 127) * 4;
  const int s0 = slot_of[2 * t], s1 = slot_of[2 * t + 1];
  const float g0 = tokp[2 * t], g1 = tokp[2 * t + 1];
  const float4 a = *(const float4*)(part + (long)s0 * NCLS + c);
  const float4 b = *(const float4*)(part + (long)s1 * NCLS + c);
  float4 r;
  r.x = g0 * a.x + g1 * b.x; r.y = g0 * a.y + g1 * b.y;
  r.z = g0 * a.z + g1 * b.z; r.w = g0 * a.w + g1 * b.w;
  *(float4*)(out + (long)t * NCLS + c) = r;
}

// ---------------- workspace layout (≈481 MB) ----------------
constexpr size_t WS_XBF   = 0;
constexpr size_t WS_W1T   = WS_XBF  + (size_t)BATCH * DIMD * 2;          // 33.5 MB
constexpr size_t WS_W2T   = WS_W1T  + (size_t)NEXP * DIMD * HID * 2;     // +67 MB
constexpr size_t WS_H     = WS_W2T  + (size_t)NEXP * HID * NCLS * 2;     // +33.5 MB
constexpr size_t WS_PART  = WS_H    + (size_t)SLOTS * HID * 2;           // +277 MB
constexpr size_t WS_TOKE  = WS_PART + (size_t)SLOTS * NCLS * 4;          // +69 MB
constexpr size_t WS_TOKP  = WS_TOKE + (size_t)BATCH * 2 * 4;
constexpr size_t WS_SLOT  = WS_TOKP + (size_t)BATCH * 2 * 4;
constexpr size_t WS_TOKOF = WS_SLOT + (size_t)BATCH * 2 * 4;
constexpr size_t WS_CNT   = WS_TOKOF + (size_t)SLOTS * 4;
constexpr size_t WS_CUR   = WS_CNT + 256;
constexpr size_t WS_OFFS  = WS_CUR + 256;
constexpr size_t WS_TILEE = WS_OFFS + 256;

extern "C" void kernel_launch(void* const* d_in, const int* in_sizes, int n_in,
                              void* d_out, int out_size, void* d_ws, size_t ws_size,
                              hipStream_t stream)
{
  const float* x      = (const float*)d_in[0];
  const float* gate_w = (const float*)d_in[1];
  const float* gate_b = (const float*)d_in[2];
  const float* w1     = (const float*)d_in[3];
  const float* b1     = (const float*)d_in[4];
  const float* ln_g   = (const float*)d_in[5];
  const float* ln_b   = (const float*)d_in[6];
  const float* w2     = (const float*)d_in[7];
  const float* b2     = (const float*)d_in[8];

  float* out    = (float*)d_out;
  float* logits = out + (size_t)BATCH * NCLS;

  uint8_t* w = (uint8_t*)d_ws;
  bf16*  xbf      = (bf16*) (w + WS_XBF);
  bf16*  w1t      = (bf16*) (w + WS_W1T);
  bf16*  w2t      = (bf16*) (w + WS_W2T);
  bf16*  hbuf     = (bf16*) (w + WS_H);
  float* part     = (float*)(w + WS_PART);
  int*   tokexp   = (int*)  (w + WS_TOKE);
  float* tokp     = (float*)(w + WS_TOKP);
  int*   slot_of  = (int*)  (w + WS_SLOT);
  int*   token_of = (int*)  (w + WS_TOKOF);
  int*   cnt      = (int*)  (w + WS_CNT);
  int*   cursor   = (int*)  (w + WS_CUR);
  int*   offs     = (int*)  (w + WS_OFFS);
  int*   tile_e   = (int*)  (w + WS_TILEE);

  // 1) router (exact fp32 logits -> d_out, top-2 + softmax probs)
  moe_router_kernel<<<BATCH / 8, 256, 0, stream>>>(x, gate_w, gate_b, logits, tokexp, tokp);

  // 2) bucketing
  moe_init_kernel<<<(SLOTS + 255) / 256, 256, 0, stream>>>(token_of, cnt, cursor);
  moe_count_kernel<<<BATCH / 256, 256, 0, stream>>>(tokexp, cnt);
  moe_scan_kernel<<<1, 32, 0, stream>>>(cnt, offs, tile_e);
  moe_fill_kernel<<<BATCH / 256, 256, 0, stream>>>(tokexp, offs, cursor, token_of, slot_of);

  // 3) precision prep: x -> bf16; w1 -> w1^T bf16 [E][H][D]; w2 -> w2^T bf16 [E][C][H]
  moe_convx_kernel<<<(size_t)BATCH * DIMD / (256 * 4), 256, 0, stream>>>(x, xbf);
  moe_transpose_conv_kernel<<<dim3(HID / 32, DIMD / 32, NEXP), 256, 0, stream>>>(w1, w1t, DIMD, HID);
  moe_transpose_conv_kernel<<<dim3(NCLS / 32, HID / 32, NEXP), 256, 0, stream>>>(w2, w2t, HID, NCLS);

  // 4) GEMM1: h = gather(x) @ w1[e] + b1[e]   (bf16 WMMA, async gathered A -> LDS)
  moe_gemm_kernel<DIMD, HID, true, true>
      <<<dim3(HID / 128, NTILES), 256, 0, stream>>>(xbf, w1t, b1, token_of, tile_e, hbuf);

  // 5) LayerNorm + exact GELU (in place, fp32 math)
  moe_ln_gelu_kernel<<<SLOTS / 8, 256, 0, stream>>>(hbuf, tile_e, ln_g, ln_b);

  // 6) GEMM2: part = h @ w2[e] + b2[e]  (fp32 partials)
  moe_gemm_kernel<HID, NCLS, false, false>
      <<<dim3(NCLS / 128, NTILES), 256, 0, stream>>>(hbuf, w2t, b2, nullptr, tile_e, part);

  // 7) combine with gate probs -> d_out
  moe_combine_kernel<<<(size_t)BATCH * NCLS / (256 * 4), 256, 0, stream>>>(part, slot_of, tokp, out);
}
